// Arch11Layer_67886253081062
// MI455X (gfx1250) — compile-verified, matrix operand order checked by reading, not run on hardware
//
#include <hip/hip_runtime.h>
#include <math.h>

// ---------------- problem constants (match reference) ----------------
#define HD   128
#define EDIM 16
#define NF_  200000
#define EI_  800000
#define NT_  20000
#define MV_  4
#define EG_  320000

typedef unsigned short u16;
typedef unsigned int   u32;
typedef unsigned long long u64;
typedef __attribute__((ext_vector_type(16))) __bf16 v16bf;
typedef __attribute__((ext_vector_type(8)))  float  v8f;

// TDM availability (device pass only)
#if defined(__AMDGCN__) && defined(__has_builtin)
#if __has_builtin(__builtin_amdgcn_tensor_load_to_lds) && __has_builtin(__builtin_amdgcn_s_wait_tensorcnt)
#define HAVE_TDM 1
#endif
#endif
#ifndef HAVE_TDM
#define HAVE_TDM 0
#endif

// ---------------- bf16 helpers ----------------
__device__ __forceinline__ u16 f2bf(float f) {
    union { float f; u32 u; } c; c.f = f;
    u32 u = c.u;
    u32 r = u + 0x7FFFu + ((u >> 16) & 1u);   // round-to-nearest-even
    return (u16)(r >> 16);
}
__device__ __forceinline__ u32 f2bf2(float lo, float hi) {
    return (u32)f2bf(lo) | ((u32)f2bf(hi) << 16);
}
__device__ __forceinline__ float bf2f(u16 s) {
    union { u32 u; float f; } c; c.u = ((u32)s) << 16;
    return c.f;
}
__device__ __forceinline__ __bf16 us2bf(u16 s) {
    union { u16 s; __bf16 h; } c; c.s = s; return c.h;
}

// ---------------- WMMA wrappers ----------------
__device__ __forceinline__ v8f wmma_bf16(v16bf a, v16bf b, v8f c) {
    return __builtin_amdgcn_wmma_f32_16x16x32_bf16(false, a, false, b, (short)0, c, false, false);
}

// Load one 16x32 bf16 fragment. `row` points to this lane's row (m for A, n for B^T),
// row-major contiguous K. Per ISA 7.12.2: lane group g=lane>>4;
// VGPR v in 0..3 holds K = k0+g*8+2v,2v+1 ; v in 4..7 holds K+16.
// The 8 pair-loads collapse into two 16B loads per lane.
__device__ __forceinline__ v16bf load_frag(const u16* row, int k0) {
    const int g8 = ((threadIdx.x >> 4) & 1) * 8;
    v16bf f;
#pragma unroll
    for (int v = 0; v < 4; ++v) {
        u32 p0 = *(const u32*)(row + k0 + g8 + 2 * v);
        u32 p1 = *(const u32*)(row + k0 + 16 + g8 + 2 * v);
        union { u32 u; u16 s[2]; } c0, c1; c0.u = p0; c1.u = p1;
        f[2 * v]     = us2bf(c0.s[0]);
        f[2 * v + 1] = us2bf(c0.s[1]);
        f[8 + 2 * v]     = us2bf(c1.s[0]);
        f[8 + 2 * v + 1] = us2bf(c1.s[1]);
    }
    return f;
}

// ---------------- TDM tile load: rows x 128 bf16 (row stride 128) -> LDS ----------------
#if HAVE_TDM
typedef u32 v4u __attribute__((ext_vector_type(4)));
typedef int v8i __attribute__((ext_vector_type(8)));
typedef int v4i __attribute__((ext_vector_type(4)));

__device__ __forceinline__ void tdm_load_bf16_tile(const u16* gsrc, u32 ldsAddr, int rows) {
    u64 ga = (u64)(uintptr_t)gsrc;
    v4u g0;
    g0[0] = 1u;                                   // count=1, user descriptor
    g0[1] = ldsAddr;                              // D#.lds_addr
    g0[2] = (u32)(ga & 0xFFFFFFFFu);              // global_addr[31:0]
    g0[3] = (u32)((ga >> 32) & 0x01FFFFFFu) | (2u << 30);  // global_addr[56:32], type=2
    const u32 d0 = 128, d1 = (u32)rows;
    v8i g1;
    g1[0] = (int)(1u << 16);                      // wg_mask=0, data_size=1 (2 bytes)
    g1[1] = (int)((d0 & 0xFFFFu) << 16);          // tensor_dim0[15:0]
    g1[2] = (int)(((d0 >> 16) & 0xFFFFu) | ((d1 & 0xFFFFu) << 16));   // dim0 hi | dim1 lo
    g1[3] = (int)(((d1 >> 16) & 0xFFFFu) | (128u << 16));             // dim1 hi | tile_dim0=128
    g1[4] = (int)(d1 & 0xFFFFu);                  // tile_dim1 = rows, tile_dim2=0
    g1[5] = (int)128;                             // tensor_dim0_stride[31:0] = 128 elems
    g1[6] = 0;
    g1[7] = 0;
    v4i z4 = {0, 0, 0, 0};
    v8i z8 = {0, 0, 0, 0, 0, 0, 0, 0};
    __builtin_amdgcn_tensor_load_to_lds(g0, g1, z4, z4, z8, 0);
}
#endif

// ---------------- small utility kernels ----------------
__global__ __launch_bounds__(256) void k_f2bf(const float* __restrict__ s, u16* __restrict__ d, int n) {
    int i = blockIdx.x * 256 + threadIdx.x;
    if (i < n) d[i] = f2bf(s[i]);
}

// GINE edge message: msg = relu(Xsrc[src] + EA@We^T + be); atomicAdd into AGG[dst].
__global__ __launch_bounds__(256) void k_gine_edge(
    const float* __restrict__ Xsrc, const float* __restrict__ EA,
    const int* __restrict__ eiSrc, const int* __restrict__ eiDst, int E,
    const float* __restrict__ We, const float* __restrict__ be,
    float* __restrict__ AGG)
{
    __shared__ float sWe[HD * EDIM];
    __shared__ float sbe[HD];
    __shared__ float sEA[2][EDIM];
    for (int i = threadIdx.x; i < HD * EDIM; i += 256) sWe[i] = We[i];
    if (threadIdx.x < HD) sbe[threadIdx.x] = be[threadIdx.x];
    __syncthreads();
    for (int base = blockIdx.x * 2; base < E; base += gridDim.x * 2) {
        if (threadIdx.x < 32) {
            int el = threadIdx.x >> 4, d = threadIdx.x & 15;
            int e = base + el;
            if (e < E) sEA[el][d] = EA[e * EDIM + d];
        }
        __syncthreads();
        int el = threadIdx.x >> 7;
        int h  = threadIdx.x & 127;
        int e  = base + el;
        if (e < E) {
            float acc = sbe[h];
#pragma unroll
            for (int d = 0; d < EDIM; ++d) acc += sEA[el][d] * sWe[h * EDIM + d];
            int s = eiSrc[e], t = eiDst[e];
            float msg = Xsrc[s * HD + h] + acc;
            msg = fmaxf(msg, 0.f);
            atomicAdd(&AGG[t * HD + h], msg);
        }
        __syncthreads();
    }
}

__global__ __launch_bounds__(256) void k_scatter_sum(
    const float* __restrict__ X, const int* __restrict__ nid,
    float* __restrict__ xsum, float* __restrict__ cnt)
{
    int i = blockIdx.x * 256 + threadIdx.x;
    if (i >= NF_ * HD) return;
    int r = i >> 7, c = i & 127;
    int nd = nid[r];
    if (nd >= 0) {
        atomicAdd(&xsum[nd * HD + c], X[i]);
        if (c == 0) atomicAdd(&cnt[nd], 1.f);
    }
}
__global__ __launch_bounds__(256) void k_div_cnt(float* __restrict__ xsum, const float* __restrict__ cnt) {
    int i = blockIdx.x * 256 + threadIdx.x;
    if (i >= NT_ * HD) return;
    xsum[i] /= fmaxf(cnt[i >> 7], 1.f);
}

// BatchNorm column stats: stats[0..127]=sum, stats[128..255]=sumsq
__global__ __launch_bounds__(256) void k_bn_stats(const float* __restrict__ X, int N, float* __restrict__ stats) {
    __shared__ float s1[256], s2[256];
    int c = threadIdx.x & 127, half = threadIdx.x >> 7;
    int rEnd = min((blockIdx.x + 1) * 1024, N);
    float s = 0.f, q = 0.f;
    for (int r = blockIdx.x * 1024 + half; r < rEnd; r += 2) {
        float v = X[r * HD + c];
        s += v; q += v * v;
    }
    s1[threadIdx.x] = s; s2[threadIdx.x] = q;
    __syncthreads();
    if (half == 0) {
        atomicAdd(&stats[c],       s1[c] + s1[c + 128]);
        atomicAdd(&stats[128 + c], s2[c] + s2[c + 128]);
    }
}
__global__ __launch_bounds__(256) void k_bn_apply(
    const float* __restrict__ X, const float* __restrict__ stats,
    const float* __restrict__ g, const float* __restrict__ b,
    int N, float* __restrict__ OUT, int accum)
{
    long long i = (long long)blockIdx.x * 256 + threadIdx.x;
    if (i >= (long long)N * HD) return;
    int c = (int)(i & 127);
    float invN = 1.f / (float)N;
    float mean = stats[c] * invN;
    float var  = stats[128 + c] * invN - mean * mean;
    float y = (X[i] - mean) * rsqrtf(var + 1e-5f) * g[c] + b[c];
    if (accum) OUT[i] += y; else OUT[i] = y;
}

// final: out = relu(acc + h2node[clamped]) * valid
__global__ __launch_bounds__(256) void k_finalize(
    float* __restrict__ OUT, const float* __restrict__ h2,
    const int* __restrict__ nid, const unsigned char* __restrict__ valid)
{
    int i = blockIdx.x * 256 + threadIdx.x;
    if (i >= NF_ * HD) return;
    int r = i >> 7, c = i & 127;
    int nd = nid[r]; nd = nd < 0 ? 0 : nd;
    float y = OUT[i] + h2[nd * HD + c];
    y = fmaxf(y, 0.f);
    OUT[i] = valid[r] ? y : 0.f;
}

// ---------------- generic WMMA GEMM: OUT = act(A @ W^T + bias) ----------------
// GMODE 0: src=row ; 1: src=max(g1[row],0) ; 2: src=g2[g1[row]]
// INBF:  X is bf16 (u16*) row-block, staged by TDM (or fallback copy)
// OUTBF: store bf16 ; ACCUM: OUT += ; ACT==1: relu
template <int GMODE, int ACCUM, int ACT, int INBF, int OUTBF>
__global__ __launch_bounds__(256) void k_gemm128(
    const void* __restrict__ Xv, const float* __restrict__ X2,
    const int* __restrict__ g1i, const int* __restrict__ g2i,
    const u16* __restrict__ W16, const float* __restrict__ bias,
    void* __restrict__ OUTv, int N)
{
    __shared__ __attribute__((aligned(16))) u16 aL[128 * HD];
    const int rb = blockIdx.x * 128;
    const int rows = min(128, N - rb);

    if constexpr (INBF) {
        const u16* Xb = (const u16*)Xv;
        u32* d32 = (u32*)aL;
        // zero tail rows (partial last block)
        for (int i = rows * 64 + threadIdx.x; i < 8192; i += 256) d32[i] = 0;
#if HAVE_TDM
        __syncthreads();
        if (threadIdx.x < 32) {   // wave 0 issues the tensor DMA
            tdm_load_bf16_tile(Xb + (size_t)rb * HD, (u32)(uintptr_t)&aL[0], rows);
            __builtin_amdgcn_s_wait_tensorcnt(0);
        }
#else
        const u32* s32 = (const u32*)(Xb + (size_t)rb * HD);
        for (int i = threadIdx.x; i < rows * 64; i += 256) d32[i] = s32[i];
#endif
    } else {
        const float* X = (const float*)Xv;
        // 128 rows x 128 cols, float4 loads, packed bf16x2 LDS stores
        for (int i = threadIdx.x; i < 128 * 32; i += 256) {
            int r = i >> 5, c4 = (i & 31) * 4;
            int gr = rb + r;
            float v0 = 0.f, v1 = 0.f, v2 = 0.f, v3 = 0.f;
            if (gr < N) {
                int src = gr;
                if constexpr (GMODE == 1) { int q = g1i[gr]; src = q < 0 ? 0 : q; }
                if constexpr (GMODE == 2) { src = g2i[g1i[gr]]; }
                const float4 x4 = *(const float4*)(X + (size_t)src * HD + c4);
                v0 = x4.x; v1 = x4.y; v2 = x4.z; v3 = x4.w;
                if (X2) {
                    const float4 y4 = *(const float4*)(X2 + (size_t)gr * HD + c4);
                    v0 += y4.x; v1 += y4.y; v2 += y4.z; v3 += y4.w;
                }
            }
            u32* d = (u32*)&aL[r * HD + c4];
            d[0] = f2bf2(v0, v1);
            d[1] = f2bf2(v2, v3);
        }
    }
    __syncthreads();

    const int wave = threadIdx.x >> 5, n15 = threadIdx.x & 15;
    const int g8 = ((threadIdx.x >> 4) & 1) * 8;
    const u16* arow = &aL[(wave * 16 + n15) * HD];
#pragma unroll 1
    for (int ct = 0; ct < 8; ++ct) {
        const u16* wrow = &W16[(ct * 16 + n15) * HD];
        v8f acc = {};
#pragma unroll
        for (int ks = 0; ks < 4; ++ks)
            acc = wmma_bf16(load_frag(arow, ks * 32), load_frag(wrow, ks * 32), acc);
        const int col = ct * 16 + n15;
        const float bs = bias ? bias[col] : 0.f;
#pragma unroll
        for (int v = 0; v < 8; ++v) {
            int r = rb + wave * 16 + v + g8;
            if (r < N) {
                float y = acc[v] + bs;
                if constexpr (ACT == 1) y = fmaxf(y, 0.f);
                if constexpr (OUTBF) {
                    ((u16*)OUTv)[(size_t)r * HD + col] = f2bf(y);
                } else if constexpr (ACCUM) {
                    ((float*)OUTv)[(size_t)r * HD + col] += y;
                } else {
                    ((float*)OUTv)[(size_t)r * HD + col] = y;
                }
            }
        }
    }
}

// ---------------- fused per-node view-attention block ----------------
__device__ __forceinline__ void ln_row(const float* x, u16* o, const float* g, const float* b) {
    float mu = 0.f;
    for (int c = 0; c < HD; ++c) mu += x[c];
    mu *= (1.f / HD);
    float var = 0.f;
    for (int c = 0; c < HD; ++c) { float d = x[c] - mu; var += d * d; }
    var *= (1.f / HD);
    float rs = rsqrtf(var + 1e-5f);
    for (int c = 0; c < HD; ++c) o[c] = f2bf((x[c] - mu) * rs * g[c] + b[c]);
}

__global__ __launch_bounds__(256) void k_view_attn(
    const float* __restrict__ h_flat, const int* __restrict__ root_flat_idx,
    const u16* __restrict__ Wqkv16, const float* __restrict__ bqkv,
    const u16* __restrict__ Wo16,   const float* __restrict__ bo,
    const u16* __restrict__ W1_16,  const float* __restrict__ fb1,
    const u16* __restrict__ W2_16,  const float* __restrict__ fb2,
    const float* __restrict__ ln1g, const float* __restrict__ ln1b,
    const float* __restrict__ ln2g, const float* __restrict__ ln2b,
    float* __restrict__ xvv)
{
    __shared__ __attribute__((aligned(16))) float xs[64 * HD];   // residual (f32)
    __shared__ __attribute__((aligned(16))) u16   ab[64 * HD];   // bf16 A staging
    __shared__ __attribute__((aligned(16))) u16   qf[64 * 512];  // qkv (stride 384) / ffn (stride 512)
    const int t = threadIdx.x;
    const int nodeBase = blockIdx.x * 16;
    const int wave = t >> 5, n15 = t & 15, g8 = ((t >> 4) & 1) * 8;

    for (int i = t; i < 64 * HD; i += 256) {
        int r = i >> 7, c = i & 127;
        int fl = root_flat_idx[(nodeBase + (r >> 2)) * MV_ + (r & 3)];
        xs[i] = h_flat[(size_t)fl * HD + c];
    }
    __syncthreads();
    if (t < 64) ln_row(&xs[t * HD], &ab[t * HD], ln1g, ln1b);
    __syncthreads();
    // qkv = LN1(x) @ Wqkv^T + bqkv (O=384) -> qf stride 384 (bf16)
    for (int tile = wave; tile < 4 * 24; tile += 8) {
        int rt = tile / 24, ct = tile % 24;
        const u16* arow = &ab[(rt * 16 + n15) * HD];
        const u16* wrow = &Wqkv16[(ct * 16 + n15) * HD];
        v8f acc = {};
#pragma unroll
        for (int ks = 0; ks < 4; ++ks)
            acc = wmma_bf16(load_frag(arow, ks * 32), load_frag(wrow, ks * 32), acc);
        int col = ct * 16 + n15;
        float bs = bqkv[col];
#pragma unroll
        for (int v = 0; v < 8; ++v)
            qf[(rt * 16 + v + g8) * 384 + col] = f2bf(acc[v] + bs);
    }
    __syncthreads();
    // 4-head attention per node (m=4 tokens, dh=32) -> ab (bf16)
    if (t < 64) {
        int node = t >> 2, head = t & 3;
        int r0 = node * 4;
        int qo = head * 32, ko = 128 + head * 32, vo = 256 + head * 32;
        float s[16];
#pragma unroll
        for (int i = 0; i < 4; ++i)
            for (int j = 0; j < 4; ++j) {
                float a = 0.f;
                for (int d = 0; d < 32; ++d)
                    a += bf2f(qf[(r0 + i) * 384 + qo + d]) * bf2f(qf[(r0 + j) * 384 + ko + d]);
                s[i * 4 + j] = a * 0.17677669529663687f;  // 32^-0.5
            }
#pragma unroll
        for (int i = 0; i < 4; ++i) {
            float mx = fmaxf(fmaxf(s[i*4], s[i*4+1]), fmaxf(s[i*4+2], s[i*4+3]));
            float den = 0.f;
            for (int j = 0; j < 4; ++j) { s[i*4+j] = expf(s[i*4+j] - mx); den += s[i*4+j]; }
            float inv = 1.f / den;
            for (int j = 0; j < 4; ++j) s[i*4+j] *= inv;
        }
        for (int d = 0; d < 32; ++d) {
            float v0 = bf2f(qf[(r0 + 0) * 384 + vo + d]);
            float v1 = bf2f(qf[(r0 + 1) * 384 + vo + d]);
            float v2 = bf2f(qf[(r0 + 2) * 384 + vo + d]);
            float v3 = bf2f(qf[(r0 + 3) * 384 + vo + d]);
#pragma unroll
            for (int i = 0; i < 4; ++i) {
                float o = s[i*4]*v0 + s[i*4+1]*v1 + s[i*4+2]*v2 + s[i*4+3]*v3;
                ab[(r0 + i) * HD + head * 32 + d] = f2bf(o);
            }
        }
    }
    __syncthreads();
    // x += attn @ Wo^T + bo
    for (int tile = wave; tile < 32; tile += 8) {
        int rt = tile >> 3, ct = tile & 7;
        const u16* arow = &ab[(rt * 16 + n15) * HD];
        const u16* wrow = &Wo16[(ct * 16 + n15) * HD];
        v8f acc = {};
#pragma unroll
        for (int ks = 0; ks < 4; ++ks)
            acc = wmma_bf16(load_frag(arow, ks * 32), load_frag(wrow, ks * 32), acc);
        int col = ct * 16 + n15;
        float bs = bo[col];
#pragma unroll
        for (int v = 0; v < 8; ++v)
            xs[(rt * 16 + v + g8) * HD + col] += acc[v] + bs;
    }
    __syncthreads();
    if (t < 64) ln_row(&xs[t * HD], &ab[t * HD], ln2g, ln2b);
    __syncthreads();
    // FFN1: gelu(LN2 @ W1^T + b1) (O=512) -> qf stride 512 (bf16)
    for (int tile = wave; tile < 4 * 32; tile += 8) {
        int rt = tile >> 5, ct = tile & 31;
        const u16* arow = &ab[(rt * 16 + n15) * HD];
        const u16* wrow = &W1_16[(ct * 16 + n15) * HD];
        v8f acc = {};
#pragma unroll
        for (int ks = 0; ks < 4; ++ks)
            acc = wmma_bf16(load_frag(arow, ks * 32), load_frag(wrow, ks * 32), acc);
        int col = ct * 16 + n15;
        float bs = fb1[col];
#pragma unroll
        for (int v = 0; v < 8; ++v) {
            float y = acc[v] + bs;
            y = 0.5f * y * (1.f + erff(y * 0.70710678118654752f));  // exact gelu
            qf[(rt * 16 + v + g8) * 512 + col] = f2bf(y);
        }
    }
    __syncthreads();
    // FFN2: x += ffn @ W2^T + b2 (K=512)
    for (int tile = wave; tile < 32; tile += 8) {
        int rt = tile >> 3, ct = tile & 7;
        const u16* arow = &qf[(rt * 16 + n15) * 512];
        const u16* wrow = &W2_16[(ct * 16 + n15) * 512];
        v8f acc = {};
#pragma unroll
        for (int ks = 0; ks < 16; ++ks)
            acc = wmma_bf16(load_frag(arow, ks * 32), load_frag(wrow, ks * 32), acc);
        int col = ct * 16 + n15;
        float bs = fb2[col];
#pragma unroll
        for (int v = 0; v < 8; ++v)
            xs[(rt * 16 + v + g8) * HD + col] += acc[v] + bs;
    }
    __syncthreads();
    // mean over m=4 views
    for (int i = t; i < 16 * HD; i += 256) {
        int node = i >> 7, c = i & 127;
        float m = 0.25f * (xs[(node*4+0)*HD + c] + xs[(node*4+1)*HD + c] +
                           xs[(node*4+2)*HD + c] + xs[(node*4+3)*HD + c]);
        xvv[(size_t)(nodeBase + node) * HD + c] = m;
    }
}

// ---------------- host launcher ----------------
extern "C" void kernel_launch(void* const* d_in, const int* in_sizes, int n_in,
                              void* d_out, int out_size, void* d_ws, size_t ws_size,
                              hipStream_t stream) {
    const float* h_flat    = (const float*)d_in[0];
    const float* ea_flat   = (const float*)d_in[1];
    const float* edge_attr = (const float*)d_in[2];
    const int*   intra_ei  = (const int*)d_in[3];
    const int*   edge_index= (const int*)d_in[4];
    const int*   node_ids  = (const int*)d_in[5];
    const int*   sub_batch = (const int*)d_in[6];
    const int*   root_idx  = (const int*)d_in[7];
    const unsigned char* valid = (const unsigned char*)d_in[8];
    const float* bskip = (const float*)d_in[12];
    const float* bvv   = (const float*)d_in[14];
    const float* bkk   = (const float*)d_in[16];
    const float* lc_We = (const float*)d_in[17]; const float* lc_be = (const float*)d_in[18];
    const float* lc_b1 = (const float*)d_in[20]; const float* lc_b2 = (const float*)d_in[22];
    const float* gc_We = (const float*)d_in[23]; const float* gc_be = (const float*)d_in[24];
    const float* gc_b1 = (const float*)d_in[26]; const float* gc_b2 = (const float*)d_in[28];
    const float* lbn_g = (const float*)d_in[29]; const float* lbn_b = (const float*)d_in[30];
    const float* gbn_g = (const float*)d_in[31]; const float* gbn_b = (const float*)d_in[32];
    const float* ln1_g = (const float*)d_in[33]; const float* ln1_b = (const float*)d_in[34];
    const float* ln2_g = (const float*)d_in[35]; const float* ln2_b = (const float*)d_in[36];
    const float* bqkv  = (const float*)d_in[38];
    const float* bo    = (const float*)d_in[40];
    const float* fb1   = (const float*)d_in[42];
    const float* fb2   = (const float*)d_in[44];
    float* out = (float*)d_out;

    // workspace layout
    char* ws = (char*)d_ws;
    size_t o = 0;
    float* agg1  = (float*)(ws + o); o += (size_t)NF_ * HD * 4;   // aggregation, then h1 raw
    u16*   tmp1b = (u16*)  (ws + o); o += (size_t)NF_ * HD * 2;   // MLP1 output (bf16)
    float* xsum  = (float*)(ws + o); o += (size_t)NT_ * HD * 4;
    float* cnt   = (float*)(ws + o); o += (size_t)NT_ * 4;
    float* agg2  = (float*)(ws + o); o += (size_t)NT_ * HD * 4;   // aggregation, then h_global raw
    u16*   tmp2b = (u16*)  (ws + o); o += (size_t)NT_ * HD * 2;
    float* h2b   = (float*)(ws + o); o += (size_t)NT_ * HD * 4;
    float* xvv   = (float*)(ws + o); o += (size_t)NT_ * HD * 4;
    float* statsL = (float*)(ws + o); o += 256 * 4;
    float* statsG = (float*)(ws + o); o += 256 * 4;
    u16* w16 = (u16*)(ws + o);
    u16* wskip = w16;             u16* wvv  = w16 + 16384;  u16* wkk  = w16 + 32768;
    u16* lcW1 = w16 + 49152;      u16* lcW2 = w16 + 65536;
    u16* gcW1 = w16 + 81920;      u16* gcW2 = w16 + 98304;
    u16* wo   = w16 + 114688;     u16* wqkv = w16 + 131072;
    u16* fW1  = w16 + 180224;     u16* fW2  = w16 + 245760;

    auto conv = [&](int idx, u16* dst, int n) {
        k_f2bf<<<(n + 255) / 256, 256, 0, stream>>>((const float*)d_in[idx], dst, n);
    };
    conv(11, wskip, 16384); conv(13, wvv, 16384); conv(15, wkk, 16384);
    conv(19, lcW1, 16384);  conv(21, lcW2, 16384);
    conv(25, gcW1, 16384);  conv(27, gcW2, 16384);
    conv(37, wqkv, 49152);  conv(39, wo, 16384);
    conv(41, fW1, 65536);   conv(43, fW2, 65536);

    (void)hipMemsetAsync(agg1, 0, (size_t)NF_ * HD * 4, stream);
    (void)hipMemsetAsync(xsum, 0, (size_t)NT_ * HD * 4, stream);
    (void)hipMemsetAsync(cnt,  0, (size_t)NT_ * 4, stream);
    (void)hipMemsetAsync(agg2, 0, (size_t)NT_ * HD * 4, stream);
    (void)hipMemsetAsync(statsL, 0, 512 * 4, stream);   // statsL + statsG contiguous

    const int gNF = (NF_ + 127) / 128, gNT = (NT_ + 127) / 128;

    // 1) out = h_flat @ Wskip^T + bskip
    k_gemm128<0,0,0,0,0><<<gNF, 256, 0, stream>>>(h_flat, nullptr, nullptr, nullptr,
                                                  wskip, bskip, out, NF_);
    // 2) local GINE messages -> agg1
    k_gine_edge<<<4096, 256, 0, stream>>>(h_flat, ea_flat, intra_ei, intra_ei + EI_, EI_,
                                          lc_We, lc_be, agg1);
    // 3) tmp1b = relu((h + agg1) @ lcW1^T + b1) (bf16); h1 = tmp1b @ lcW2^T + b2 -> agg1
    k_gemm128<0,0,1,0,1><<<gNF, 256, 0, stream>>>(h_flat, agg1, nullptr, nullptr,
                                                  lcW1, lc_b1, tmp1b, NF_);
    k_gemm128<0,0,0,1,0><<<gNF, 256, 0, stream>>>(tmp1b, nullptr, nullptr, nullptr,
                                                  lcW2, lc_b2, agg1, NF_);
    // 4) BN(local), accumulate into out
    k_bn_stats<<<(NF_ + 1023) / 1024, 256, 0, stream>>>(agg1, NF_, statsL);
    k_bn_apply<<<(NF_ * HD + 255) / 256, 256, 0, stream>>>(agg1, statsL, lbn_g, lbn_b,
                                                           NF_, out, 1);
    // 5) scatter-mean to global nodes
    k_scatter_sum<<<(NF_ * HD + 255) / 256, 256, 0, stream>>>(h_flat, node_ids, xsum, cnt);
    k_div_cnt<<<(NT_ * HD + 255) / 256, 256, 0, stream>>>(xsum, cnt);
    // 6) global GINE
    k_gine_edge<<<4096, 256, 0, stream>>>(xsum, edge_attr, edge_index, edge_index + EG_, EG_,
                                          gc_We, gc_be, agg2);
    k_gemm128<0,0,1,0,1><<<gNT, 256, 0, stream>>>(xsum, agg2, nullptr, nullptr,
                                                  gcW1, gc_b1, tmp2b, NT_);
    k_gemm128<0,0,0,1,0><<<gNT, 256, 0, stream>>>(tmp2b, nullptr, nullptr, nullptr,
                                                  gcW2, gc_b2, agg2, NT_);
    // 7) BN(global) -> h2b
    k_bn_stats<<<(NT_ + 1023) / 1024, 256, 0, stream>>>(agg2, NT_, statsG);
    k_bn_apply<<<(NT_ * HD + 255) / 256, 256, 0, stream>>>(agg2, statsG, gbn_g, gbn_b,
                                                           NT_, h2b, 0);
    // 8) fused view-attention -> xvv
    k_view_attn<<<NT_ / 16, 256, 0, stream>>>(h_flat, root_idx,
                                              wqkv, bqkv, wo, bo, fW1, fb1, fW2, fb2,
                                              ln1_g, ln1_b, ln2_g, ln2_b, xvv);
    // 9) out += xvv[clamped(node_ids)] @ Wvv^T + bvv
    k_gemm128<1,1,0,0,0><<<gNF, 256, 0, stream>>>(xvv, nullptr, node_ids, nullptr,
                                                  wvv, bvv, out, NF_);
    // 10) out += h_flat[root_idx[sub_batch]] @ Wkk^T + bkk
    k_gemm128<2,1,0,0,0><<<gNF, 256, 0, stream>>>(h_flat, nullptr, sub_batch, root_idx,
                                                  wkk, bkk, out, NF_);
    // 11) out = relu(out + h2b[clamped]) * valid
    k_finalize<<<(NF_ * HD + 255) / 256, 256, 0, stream>>>(out, h2b, node_ids, valid);
}
